// FourrierKernelAttention_54537494724815
// MI455X (gfx1250) — matrix operand order repaired and checked by reading, not compiled, not run on hardware
//
#include <hip/hip_runtime.h>
#include <hip/hip_bf16.h>
#include <math.h>

// ---------------------------------------------------------------------------
// FourierKernelAttention on gfx1250 (MI455X, wave32, WMMA).
//   N=2, L=1024, H=8, d=64, D=512, feature dim F=2d=128 (cos|sin), P=1.
//   Chunked causal linear attention; all GEMM-shaped work on
//   v_wmma_f32_16x16x32_bf16 (f32 accumulate); chunk operands staged to LDS
//   via GLOBAL_LOAD_ASYNC_TO_LDS_B128 when the toolchain exposes it.
// ---------------------------------------------------------------------------

typedef __attribute__((ext_vector_type(16))) __bf16 bf16x16;
typedef __attribute__((ext_vector_type(8)))  float  f32x8;
typedef int fka_v4i __attribute__((vector_size(16)));

#define NB   2
#define LSEQ 1024
#define HH   8
#define DH   64
#define DD   512
#define FF   128   // feature dim (cos | sin halves)
#define CC   64    // scan chunk size

#if defined(__gfx1250__) &&                                            \
    __has_builtin(__builtin_amdgcn_global_load_async_to_lds_b128) &&   \
    __has_builtin(__builtin_amdgcn_s_wait_asynccnt)
#define FKA_ASYNC_LDS 1
typedef __attribute__((address_space(1))) fka_v4i* fka_gp;
typedef __attribute__((address_space(3))) fka_v4i* fka_lp;
#endif

static __device__ __forceinline__ f32x8 wmma_bf16(bf16x16 a, bf16x16 b, f32x8 c) {
  // (neg_a, A, neg_b, B, c_mod, C, reuse_a, reuse_b)
  return __builtin_amdgcn_wmma_f32_16x16x32_bf16(false, a, false, b, (short)0, c,
                                                 false, false);
}

// Copy `bytes` (multiple of 4KB) from global to LDS, 16B per lane per sweep.
static __device__ __forceinline__ void stage_to_lds(const __bf16* g, __bf16* l,
                                                    int bytes) {
  const int tid = threadIdx.x;
#if defined(FKA_ASYNC_LDS)
  for (int s = tid * 16; s < bytes; s += 256 * 16) {
    __builtin_amdgcn_global_load_async_to_lds_b128(
        (fka_gp)((const char*)g + s), (fka_lp)((char*)l + s), 0, 0);
  }
#else
  for (int s = tid * 16; s < bytes; s += 256 * 16) {
    uint4 d = *(const uint4*)((const char*)g + s);
    *(uint4*)((char*)l + s) = d;
  }
#endif
}

static __device__ __forceinline__ void stage_wait() {
#if defined(FKA_ASYNC_LDS)
  __builtin_amdgcn_s_wait_asynccnt(0);
#endif
  __syncthreads();
}

// A fragment: 16x32 bf16 tile of row-major [ld] matrix at (row0, col0).
// Lane (m = lane&15, upper half handles K+8): two contiguous 8-elem runs.
static __device__ __forceinline__ bf16x16 load_a_bf16(const __bf16* p, int ld,
                                                      int row0, int col0) {
  const int lane = threadIdx.x & 31;
  const int m = lane & 15;
  const int up = (lane >> 4) << 3;          // 0 or 8
  const __bf16* r = p + (size_t)(row0 + m) * ld + col0 + up;
  bf16x16 a;
#pragma unroll
  for (int i = 0; i < 8; ++i) { a[i] = r[i]; a[8 + i] = r[16 + i]; }
  return a;
}

// Same but source matrix is f32 (convert to bf16 on the fly).
static __device__ __forceinline__ bf16x16 load_a_f32(const float* p, int ld,
                                                     int row0, int col0) {
  const int lane = threadIdx.x & 31;
  const int m = lane & 15;
  const int up = (lane >> 4) << 3;
  const float* r = p + (size_t)(row0 + m) * ld + col0 + up;
  bf16x16 a;
#pragma unroll
  for (int i = 0; i < 8; ++i) { a[i] = (__bf16)r[i]; a[8 + i] = (__bf16)r[16 + i]; }
  return a;
}

// A fragment of M^T: A[m][k] = p[(col0+k)*ld + row0+m]   (LDS gather)
static __device__ __forceinline__ bf16x16 load_a_bf16_T(const __bf16* p, int ld,
                                                        int row0, int col0) {
  const int lane = threadIdx.x & 31;
  const int m = lane & 15;
  const int up = (lane >> 4) << 3;
  bf16x16 a;
#pragma unroll
  for (int i = 0; i < 8; ++i) {
    a[i]     = p[(size_t)(col0 + up + i) * ld + row0 + m];
    a[8 + i] = p[(size_t)(col0 + 16 + up + i) * ld + row0 + m];
  }
  return a;
}

// B fragment: 32x16 tile, lane holds col (col0+lane%16), K rows up+i and 16+up+i.
static __device__ __forceinline__ bf16x16 load_b_bf16(const __bf16* p, int ld,
                                                      int row0, int col0) {
  const int lane = threadIdx.x & 31;
  const int up = (lane >> 4) << 3;
  const __bf16* r = p + (size_t)(row0 + up) * ld + col0 + (lane & 15);
  bf16x16 b;
#pragma unroll
  for (int i = 0; i < 8; ++i) { b[i] = r[(size_t)i * ld]; b[8 + i] = r[(size_t)(16 + i) * ld]; }
  return b;
}

// B fragment of M^T: B[k][n] = p[(col0+n)*ld + row0+k], n = lane%16, k = up+i / 16+up+i.
static __device__ __forceinline__ bf16x16 load_b_bf16_T(const __bf16* p, int ld,
                                                        int row0, int col0) {
  const int lane = threadIdx.x & 31;
  const int up = (lane >> 4) << 3;
  const __bf16* r = p + (size_t)(col0 + (lane & 15)) * ld + row0 + up;
  bf16x16 b;
#pragma unroll
  for (int i = 0; i < 8; ++i) { b[i] = r[i]; b[8 + i] = r[16 + i]; }
  return b;
}

static __device__ __forceinline__ float softplus_f(float x) {
  return (x > 20.f) ? x : log1pf(expf(x));
}

// ---------------------------------------------------------------------------
// Kernel 0: transpose + bf16-convert the three 512x512 weights:
//   Wt[m][d][e] = W_m[e][d],  so WMMA B operands are contiguous.
// ---------------------------------------------------------------------------
__global__ __launch_bounds__(256)
void fka_transpose_w(const float* __restrict__ Wq, const float* __restrict__ Wk,
                     const float* __restrict__ Wv, __bf16* __restrict__ Wt) {
  const int idx = blockIdx.x * blockDim.x + threadIdx.x;   // 3*512*512 threads
  const int m = idx >> 18;
  const int r = idx & ((1 << 18) - 1);
  const int e = r >> 9, dd = r & 511;
  const float* W = (m == 0) ? Wq : ((m == 1) ? Wk : Wv);
  Wt[(size_t)m * DD * DD + (size_t)dd * DD + e] = (__bf16)W[(size_t)e * DD + dd];
}

// ---------------------------------------------------------------------------
// Kernel 1: fused projection GEMM + feature map.
//   q/k/v projections via WMMA; epilogue applies softplus and Fourier
//   features, writes bf16 phiQ/phiK (N,H,L,128) and V (N,H,L,64).
// One block = 64 rows of the (N*L) x 512 output, 8 waves.
// ---------------------------------------------------------------------------
__global__ __launch_bounds__(256)
void fka_proj_features(const float* __restrict__ query,
                       const float* __restrict__ key_seq,
                       const __bf16* __restrict__ Wt,
                       const float* __restrict__ coeff,
                       const float* __restrict__ pw,
                       const float* __restrict__ pb,
                       __bf16* __restrict__ phiQ, __bf16* __restrict__ phiK,
                       __bf16* __restrict__ Vt) {
  const int wave = threadIdx.x >> 5;
  const int lane = threadIdx.x & 31;
  const int m0 = blockIdx.x * 64;           // 32 blocks cover 2048 rows
  const int Mofs = (lane >> 4) << 3;
  const int Nn = lane & 15;

  for (int mat = 0; mat < 3; ++mat) {       // 0:q 1:k 2:v
    const float* X = (mat == 0) ? query : key_seq;
    const __bf16* W = Wt + (size_t)mat * DD * DD;
#pragma unroll
    for (int rt = 0; rt < 4; ++rt) {
      const int row0 = m0 + rt * 16;
      for (int ccol = 0; ccol < 4; ++ccol) {
        const int e0 = (wave + 8 * ccol) * 16;     // column tile
        f32x8 acc = {};
#pragma unroll 4
        for (int kt = 0; kt < 16; ++kt) {
          bf16x16 a = load_a_f32(X, DD, row0, kt * 32);
          bf16x16 b = load_b_bf16(W, DD, kt * 32, e0);
          acc = wmma_bf16(a, b, acc);
        }
#pragma unroll
        for (int r = 0; r < 8; ++r) {
          const int grow = row0 + r + Mofs;
          const int n = grow >> 10, l = grow & 1023;
          const int e = e0 + Nn;                  // 0..511
          const int h = e >> 6, j = e & 63;
          const float val = acc[r];
          const size_t rowNHL = (size_t)(n * HH + h) * LSEQ + l;
          if (mat == 2) {
            Vt[rowNHL * DH + j] = (__bf16)val;
          } else {
            const float sp = softplus_f(val);
            const float ang = (mat == 0) ? ((float)l * pw[e] + pb[e])
                                         : ((float)l * pw[e]);
            float sv, cv;
            sincosf(ang, &sv, &cv);
            const float scale = (mat == 0) ? coeff[e] : 1.0f;
            __bf16* dst = ((mat == 0) ? phiQ : phiK) + rowNHL * FF;
            dst[j]      = (__bf16)(sp * cv * scale);
            dst[DH + j] = (__bf16)(sp * sv * scale);
          }
        }
      }
    }
  }
}

// ---------------------------------------------------------------------------
// Kernel 2: chunked causal scan, one block per (n,h), 8 waves.
//   State S (128x64 f32) lives in persistent WMMA accumulators: wave w owns
//   feature rows [16w, 16w+16). A bf16 mirror of S sits in LDS as the B
//   operand for the inter-chunk product. Per chunk c (CC=64 rows):
//     stage : async-copy phiK/V chunk into LDS (overlaps phase 0)
//     phase0: den[t]  = phiQ[t] . z
//     phase1: A = phiQ phiK^T (WMMA), causal mask, row-sums -> den, A -> LDS
//     phase2: O = A@V + phiQ@S_prev (WMMA), out = O / den
//     phase3: Sreg += phiK^T @ V (WMMA accumulate), refresh bf16 mirror,
//             z += colsum(phiK)
// ---------------------------------------------------------------------------
__global__ __launch_bounds__(256)
void fka_scan(const __bf16* __restrict__ phiQ, const __bf16* __restrict__ phiK,
              const __bf16* __restrict__ Vt, float* __restrict__ out) {
  __shared__ __bf16 sSb[FF * DH];    // 16KB  bf16 state mirror (WMMA B operand)
  __shared__ __bf16 sA[CC * CC];     //  8KB  masked chunk scores
  __shared__ __bf16 sK[CC * FF];     // 16KB  staged phiK chunk
  __shared__ __bf16 sV[CC * DH];     //  8KB  staged V chunk
  __shared__ float  sZ[FF];
  __shared__ float  sDen[CC];

  const int nh = blockIdx.x;                 // 0..15
  const int n = nh >> 3, h = nh & 7;
  const __bf16* pQ = phiQ + (size_t)nh * LSEQ * FF;
  const __bf16* pK = phiK + (size_t)nh * LSEQ * FF;
  const __bf16* pV = Vt   + (size_t)nh * LSEQ * DH;

  const int tid = threadIdx.x;
  const int wave = tid >> 5, lane = tid & 31;
  const int Mofs = (lane >> 4) << 3;
  const int Nn = lane & 15;
  const int rt = wave >> 1;                  // output/score row tile 0..3
  const int ft = wave;                       // state feature tile 0..7

  f32x8 Sreg[4];                             // persistent state accumulators
#pragma unroll
  for (int ce = 0; ce < 4; ++ce) Sreg[ce] = f32x8{};

  for (int i = tid; i < FF * DH; i += 256) sSb[i] = (__bf16)0.f;
  if (tid < FF) sZ[tid] = 0.f;
  __syncthreads();

  for (int c = 0; c < LSEQ / CC; ++c) {
    const int base = c * CC;

    // ---- stage phiK/V chunk into LDS (async; overlaps phase 0) ----------
    stage_to_lds(pK + (size_t)base * FF, sK, CC * FF * 2);
    stage_to_lds(pV + (size_t)base * DH, sV, CC * DH * 2);

    // phiQ A-fragments shared by phases 1 and 2 (global, contiguous runs)
    bf16x16 aQ[4];
#pragma unroll
    for (int kf = 0; kf < 4; ++kf)
      aQ[kf] = load_a_bf16(pQ, FF, base + rt * 16, kf * 32);

    // ---- phase 0: inter-chunk denominator base --------------------------
    if (tid < CC) {
      float dsum = 0.f;
      const __bf16* qrow = pQ + (size_t)(base + tid) * FF;
#pragma unroll 8
      for (int f = 0; f < FF; ++f) dsum += (float)qrow[f] * sZ[f];
      sDen[tid] = dsum;
    }
    stage_wait();                            // asynccnt drain + barrier

    // ---- phase 1: masked intra-chunk scores -----------------------------
#pragma unroll
    for (int cc2 = 0; cc2 < 2; ++cc2) {
      const int ct = (wave & 1) + 2 * cc2;   // col tile 0..3
      if (ct > rt) {                         // fully above diagonal: zeros
#pragma unroll
        for (int r = 0; r < 8; ++r)
          sA[(rt * 16 + r + Mofs) * CC + ct * 16 + Nn] = (__bf16)0.f;
        continue;
      }
      f32x8 acc = {};
#pragma unroll
      for (int kf = 0; kf < 4; ++kf) {
        bf16x16 b = load_b_bf16_T(sK, FF, kf * 32, ct * 16);
        acc = wmma_bf16(aQ[kf], b, acc);
      }
#pragma unroll
      for (int r = 0; r < 8; ++r) {
        const int tl = rt * 16 + r + Mofs;
        const int ul = ct * 16 + Nn;
        const float v = (ct < rt || ul <= tl) ? acc[r] : 0.f;
        atomicAdd(&sDen[tl], v);             // ds_add_f32 row-sum
        sA[tl * CC + ul] = (__bf16)v;
      }
    }
    __syncthreads();

    // ---- phase 2: O = A@V + phiQ@S_prev ; out = O / den -----------------
#pragma unroll
    for (int cc2 = 0; cc2 < 2; ++cc2) {
      const int e0 = ((wave & 1) + 2 * cc2) * 16;
      f32x8 acc = {};
#pragma unroll
      for (int ku = 0; ku < 2; ++ku) {       // intra-chunk
        bf16x16 a = load_a_bf16(sA, CC, rt * 16, ku * 32);
        bf16x16 b = load_b_bf16(sV, DH, ku * 32, e0);
        acc = wmma_bf16(a, b, acc);
      }
#pragma unroll
      for (int kf = 0; kf < 4; ++kf) {       // inter-chunk via state mirror
        bf16x16 b = load_b_bf16(sSb, DH, kf * 32, e0);
        acc = wmma_bf16(aQ[kf], b, acc);
      }
#pragma unroll
      for (int r = 0; r < 8; ++r) {
        const int tl = rt * 16 + r + Mofs;
        const int l = base + tl;
        out[((size_t)(n * LSEQ + l) * HH + h) * DH + e0 + Nn] =
            acc[r] / sDen[tl];
      }
    }
    __syncthreads();

    // ---- phase 3: state update (pure WMMA accumulate) -------------------
    {
      bf16x16 aT[2];
      aT[0] = load_a_bf16_T(sK, FF, ft * 16, 0);
      aT[1] = load_a_bf16_T(sK, FF, ft * 16, 32);
#pragma unroll
      for (int ce = 0; ce < 4; ++ce) {
#pragma unroll
        for (int ku = 0; ku < 2; ++ku) {
          bf16x16 b = load_b_bf16(sV, DH, ku * 32, ce * 16);
          Sreg[ce] = wmma_bf16(aT[ku], b, Sreg[ce]);
        }
#pragma unroll
        for (int r = 0; r < 8; ++r)
          sSb[(ft * 16 + r + Mofs) * DH + ce * 16 + Nn] = (__bf16)Sreg[ce][r];
      }
      if (tid < FF) {                        // z += column sums of phiK
        float zz = 0.f;
#pragma unroll 8
        for (int u = 0; u < CC; ++u) zz += (float)sK[u * FF + tid];
        sZ[tid] += zz;
      }
    }
    __syncthreads();
  }
}

// ---------------------------------------------------------------------------
extern "C" void kernel_launch(void* const* d_in, const int* in_sizes, int n_in,
                              void* d_out, int out_size, void* d_ws, size_t ws_size,
                              hipStream_t stream) {
  (void)in_sizes; (void)n_in; (void)out_size; (void)ws_size;
  const float* query  = (const float*)d_in[0];
  const float* keyseq = (const float*)d_in[1];
  const float* Wq     = (const float*)d_in[2];
  const float* Wk     = (const float*)d_in[3];
  const float* Wv     = (const float*)d_in[4];
  const float* coeff  = (const float*)d_in[5];
  const float* pw     = (const float*)d_in[6];   // (H,d,1) flat = 512
  const float* pb     = (const float*)d_in[7];
  float* out = (float*)d_out;

  char* ws = (char*)d_ws;
  __bf16* Wt   = (__bf16*)(ws);                      // 3*512*512*2 = 1.5 MB
  __bf16* phiQ = (__bf16*)(ws + 1572864);            // 2*8*1024*128*2 = 4 MB
  __bf16* phiK = (__bf16*)(ws + 5767168);            // 4 MB
  __bf16* Vt   = (__bf16*)(ws + 9961472);            // 2*8*1024*64*2 = 2 MB

  fka_transpose_w<<<3 * DD * DD / 256, 256, 0, stream>>>(Wq, Wk, Wv, Wt);
  fka_proj_features<<<(NB * LSEQ) / 64, 256, 0, stream>>>(
      query, keyseq, Wt, coeff, pw, pb, phiQ, phiK, Vt);
  fka_scan<<<NB * HH, 256, 0, stream>>>(phiQ, phiK, Vt, out);
}
